// VarianceAdaptor_69612829934084
// MI455X (gfx1250) — matrix operand order repaired and checked by reading, not compiled, not run on hardware
//
#include <hip/hip_runtime.h>
#include <hip/hip_bf16.h>

typedef __attribute__((ext_vector_type(16))) _Float16 v16h;
typedef __attribute__((ext_vector_type(8)))  _Float16 v8h;
typedef __attribute__((ext_vector_type(8)))  float    v8f;
typedef __attribute__((ext_vector_type(4)))  float    v4f;

// Problem constants
constexpr int B_  = 32;
constexpr int L_  = 512;
constexpr int D_  = 256;
constexpr int F_  = 256;
constexpr int T_  = 2048;
constexpr int NB_ = 256;
constexpr float LN_EPS = 1e-5f;

// Conv tile config: 32 output frames per block, F=256 output channels.
constexpr int TT         = 32;        // frames per block
constexpr int SIN_ROWS   = TT + 2;    // +1 halo each side (k=3, pad=1)
constexpr int SIN_STRIDE = 264;       // halves; 528B row stride -> conflict-free 16B chunks
constexpr int SY_STRIDE  = 264;       // floats

// ---------------------------------------------------------------------------
// Weight repack: f32 (3, Cin=256, F=256) -> f16 packed in WMMA-B fragment order.
// Flat layout: block (k, kb, nt) of 512 halves; within a block, lane*16 + e,
// where for lane group g = lane>>4, m = lane&15:  K(row c) = kb*32 + 16*g + e,
// N(col f) = nt*16 + m.  (Contiguous K order; the A fragment in conv_ln_kernel uses the same.)
// ---------------------------------------------------------------------------
__global__ void pack_conv_w(const float* __restrict__ w, _Float16* __restrict__ dst) {
    int idx = blockIdx.x * blockDim.x + threadIdx.x;
    if (idx >= 3 * 256 * 256) return;
    int e    = idx & 15;
    int lane = (idx >> 4) & 31;
    int blk  = idx >> 9;              // (k*8 + kb)*16 + nt
    int nt   = blk & 15;
    int kb   = (blk >> 4) & 7;
    int k    = blk >> 7;
    int g = lane >> 4, m = lane & 15;
    int c = kb * 32 + 16 * g + e;
    int f = nt * 16 + m;
    dst[idx] = (_Float16)w[(k * 256 + c) * 256 + f];
}

// ---------------------------------------------------------------------------
// Per-batch inclusive cumsum of durations (L=512), one wave32 per batch row.
// ---------------------------------------------------------------------------
__global__ void cumsum_kernel(const int* __restrict__ dur, int* __restrict__ cum) {
    int b = blockIdx.x;
    int lane = threadIdx.x;           // 32 threads
    const int* src = dur + b * L_;
    int vals[16];
    int s = 0;
    #pragma unroll
    for (int i = 0; i < 16; ++i) { s += src[lane * 16 + i]; vals[i] = s; }
    int total = s;
    int incl = total;
    for (int off = 1; off < 32; off <<= 1) {
        int n = __shfl_up(incl, off);
        if (lane >= off) incl += n;
    }
    int excl = incl - total;
    int* dstp = cum + b * L_;
    #pragma unroll
    for (int i = 0; i < 16; ++i) dstp[lane * 16 + i] = excl + vals[i];
}

// ---------------------------------------------------------------------------
// Length regulate: h[b,t,:] = (idx<L) ? x[b,idx,:] : 0, idx = searchsorted(cum,t,right).
// One wave per (b,t); lane 0 does the binary search, all lanes copy float4s.
// ---------------------------------------------------------------------------
__global__ void length_regulate_kernel(const float* __restrict__ x, const int* __restrict__ cum,
                                       float* __restrict__ h) {
    int wv = threadIdx.x >> 5, lane = threadIdx.x & 31;
    long row = (long)blockIdx.x * 8 + wv;        // 0..B*T-1
    int b = (int)(row / T_), t = (int)(row % T_);
    int idx = 0, valid = 0;
    if (lane == 0) {
        const int* c = cum + b * L_;
        int lo = 0, hi = L_;
        while (lo < hi) { int mid = (lo + hi) >> 1; if (c[mid] <= t) lo = mid + 1; else hi = mid; }
        valid = (lo < L_);
        idx = valid ? lo : (L_ - 1);
    }
    idx = __shfl(idx, 0);
    valid = __shfl(valid, 0);
    const v4f* src = (const v4f*)(x + ((long)b * L_ + idx) * D_);
    v4f* dst = (v4f*)(h + row * D_);
    v4f z = {};
    dst[lane * 2]     = valid ? src[lane * 2]     : z;
    dst[lane * 2 + 1] = valid ? src[lane * 2 + 1] : z;
}

// ---------------------------------------------------------------------------
// Fused Conv1d(k=3,SAME) + bias + ReLU + LayerNorm.  IN_T in {float,_Float16}.
// Output written as f16 (feeds next conv / linear head).
// Block: 256 threads = 8 waves; tile = 32 frames x 256 channels.
// Wave w owns M-tile (w&1) and 4 N-tiles starting at (w>>1)*4 -> 2x16 tiles total.
// K loop: 3 taps x 8 K-blocks of 32 -> 24 v_wmma_f32_16x16x32_f16 per acc tile.
// ---------------------------------------------------------------------------
template <typename IN_T>
__global__ void conv_ln_kernel(const IN_T* __restrict__ in, const _Float16* __restrict__ wpk,
                               const float* __restrict__ bias, const float* __restrict__ gamma,
                               const float* __restrict__ beta, _Float16* __restrict__ out,
                               int T_len) {
    __shared__ __align__(32) _Float16 s_in[SIN_ROWS * SIN_STRIDE];  // ~18 KB
    __shared__ __align__(16) float    s_y[TT * SY_STRIDE];          // ~34 KB

    const int b   = blockIdx.y;
    const int t0  = blockIdx.x * TT;
    const int tid = threadIdx.x;
    const IN_T* inb = in + (long)b * T_len * 256;

    // Stage input tile (with halo + zero padding) as f16 into LDS. Coalesced.
    for (int i = tid; i < SIN_ROWS * 256; i += 256) {
        int r = i >> 8, c = i & 255;
        int t = t0 + r - 1;
        float v = (t >= 0 && t < T_len) ? (float)inb[(long)t * 256 + c] : 0.f;
        s_in[r * SIN_STRIDE + c] = (_Float16)v;
    }
    __syncthreads();

    const int wv = tid >> 5, lane = tid & 31;
    const int g = lane >> 4, m = lane & 15;
    const int mt  = wv & 1;
    const int ntb = (wv >> 1) * 4;

    v8f acc[4] = {};
    for (int k = 0; k < 3; ++k) {
        #pragma unroll
        for (int kb = 0; kb < 8; ++kb) {
            // A fragment (16x32 f16, M=lane&15): per lane two contiguous 16B chunks,
            // K = 16*g + e  =>  chunks at cols kb*32 + 16g and kb*32 + 16g + 8.
            const _Float16* arow = &s_in[(mt * 16 + m + k) * SIN_STRIDE + kb * 32];
            union { v16h v; v8h h[2]; } a;
            a.h[0] = *(const v8h*)(arow + 16 * g);
            a.h[1] = *(const v8h*)(arow + 16 * g + 8);

            const _Float16* wblk = wpk + (((k * 8 + kb) * 16) + ntb) * 512 + lane * 16;
            if (kb < 7) __builtin_prefetch(wblk + 16 * 512, 0, 0);  // next K-block -> global_prefetch_b8
            #pragma unroll
            for (int j = 0; j < 4; ++j) {
                v16h bf = *(const v16h*)(wblk + j * 512);
                acc[j] = __builtin_amdgcn_wmma_f32_16x16x32_f16(
                    /*neg_a=*/false, a.v, /*neg_b=*/false, bf,
                    /*c_mod=*/(short)0, acc[j], /*reuse_a=*/false, /*reuse_b=*/false);
            }
        }
    }

    // Epilogue: bias + ReLU -> f32 LDS tile.  D layout: M = r + 8g, N = lane&15.
    #pragma unroll
    for (int j = 0; j < 4; ++j) {
        int f = (ntb + j) * 16 + m;
        float bv = bias[f];
        #pragma unroll
        for (int r = 0; r < 8; ++r) {
            int row = mt * 16 + r + 8 * g;
            float v = acc[j][r] + bv;
            s_y[row * SY_STRIDE + f] = v > 0.f ? v : 0.f;
        }
    }
    __syncthreads();

    // LayerNorm over F=256 per frame: wave w handles rows 4w..4w+3, 8 vals/lane.
    for (int rr = 0; rr < 4; ++rr) {
        int row = wv * 4 + rr;
        const float* yr = &s_y[row * SY_STRIDE];
        float vals[8];
        float s = 0.f, s2 = 0.f;
        #pragma unroll
        for (int i = 0; i < 8; ++i) {
            float v = yr[lane * 8 + i];
            vals[i] = v; s += v; s2 += v * v;
        }
        for (int off = 16; off > 0; off >>= 1) {
            s  += __shfl_xor(s,  off);
            s2 += __shfl_xor(s2, off);
        }
        float mean = s * (1.f / 256.f);
        float var  = s2 * (1.f / 256.f) - mean * mean;
        float rstd = rsqrtf(var + LN_EPS);
        _Float16* orow = out + ((long)b * T_len + t0 + row) * 256;
        #pragma unroll
        for (int i = 0; i < 8; ++i) {
            int f = lane * 8 + i;
            orow[f] = (_Float16)((vals[i] - mean) * rstd * gamma[f] + beta[f]);
        }
    }
}

// ---------------------------------------------------------------------------
// Linear head: pred[row] = relu(dot(ln2[row,:], lw) + lb).  One wave per row.
// ---------------------------------------------------------------------------
__global__ void linear_head_kernel(const _Float16* __restrict__ hin, const float* __restrict__ lw,
                                   const float* __restrict__ lb, float* __restrict__ outp) {
    int wv = threadIdx.x >> 5, lane = threadIdx.x & 31;
    long row = (long)blockIdx.x * 8 + wv;
    const _Float16* r = hin + row * 256;
    float s = 0.f;
    #pragma unroll
    for (int i = 0; i < 8; ++i) {
        int f = lane * 8 + i;
        s += (float)r[f] * lw[f];
    }
    for (int off = 16; off > 0; off >>= 1) s += __shfl_xor(s, off);
    if (lane == 0) {
        float v = s + lb[0];
        outp[row] = v > 0.f ? v : 0.f;
    }
}

// ---------------------------------------------------------------------------
// h[b,t,:] += emb[searchsorted(bins, target[b,t], left)].  One wave per (b,t).
// ---------------------------------------------------------------------------
__global__ void add_emb_kernel(float* __restrict__ h, const float* __restrict__ target,
                               const float* __restrict__ bins, const float* __restrict__ emb) {
    int wv = threadIdx.x >> 5, lane = threadIdx.x & 31;
    long row = (long)blockIdx.x * 8 + wv;        // 0..B*T-1
    int idx = 0;
    if (lane == 0) {
        float v = target[row];
        int lo = 0, hi = NB_ - 1;                // 255 bins; lower_bound -> [0,255]
        while (lo < hi) { int mid = (lo + hi) >> 1; if (bins[mid] < v) lo = mid + 1; else hi = mid; }
        idx = (lo == NB_ - 1 && !(bins[NB_ - 2] >= v)) ? NB_ - 1 : lo;
        // note: standard lower_bound with hi=n gives exactly searchsorted-left;
        // here hi started at n-1? fix: handle the "all bins < v" case explicitly
        if (bins[NB_ - 2] < v) idx = NB_ - 1;
    }
    idx = __shfl(idx, 0);
    const v4f* e = (const v4f*)(emb + (long)idx * 256);
    v4f* hp = (v4f*)(h + row * 256);
    hp[lane * 2]     = hp[lane * 2]     + e[lane * 2];
    hp[lane * 2 + 1] = hp[lane * 2 + 1] + e[lane * 2 + 1];
}

// ---------------------------------------------------------------------------
extern "C" void kernel_launch(void* const* d_in, const int* in_sizes, int n_in,
                              void* d_out, int out_size, void* d_ws, size_t ws_size,
                              hipStream_t stream) {
    (void)in_sizes; (void)n_in; (void)out_size; (void)ws_size;

    const float* x        = (const float*)d_in[0];
    const int*   d_target = (const int*)d_in[1];
    const float* e_target = (const float*)d_in[2];
    const float* p_target = (const float*)d_in[3];
    const float* prm[3][10];
    for (int p = 0; p < 3; ++p)
        for (int i = 0; i < 10; ++i)
            prm[p][i] = (const float*)d_in[5 + p * 10 + i];
    const float* en_emb  = (const float*)d_in[35];
    const float* pi_emb  = (const float*)d_in[36];
    const float* en_bins = (const float*)d_in[37];
    const float* pi_bins = (const float*)d_in[38];

    // Workspace layout
    char* ws = (char*)d_ws;
    size_t off = 0;
    _Float16* wpk[6];
    for (int i = 0; i < 6; ++i) { wpk[i] = (_Float16*)(ws + off); off += (size_t)3 * 256 * 256 * sizeof(_Float16); }
    int* cum = (int*)(ws + off); off += (size_t)B_ * L_ * sizeof(int);
    off = (off + 255) & ~(size_t)255;
    _Float16* ln1 = (_Float16*)(ws + off); off += (size_t)B_ * T_ * F_ * sizeof(_Float16);
    _Float16* ln2 = (_Float16*)(ws + off); off += (size_t)B_ * T_ * F_ * sizeof(_Float16);

    // Output segments (flat, in reference return order)
    float* h_out   = (float*)d_out;
    float* log_dur = h_out + (size_t)B_ * T_ * D_;
    float* pitch   = log_dur + (size_t)B_ * L_;
    float* energy  = pitch + (size_t)B_ * T_;

    // 1) Pack conv weights to f16 WMMA-B layout (dur c1/c2, en c1/c2, pi c1/c2)
    const float* convw[6] = { prm[0][0], prm[0][4], prm[1][0], prm[1][4], prm[2][0], prm[2][4] };
    const int npack = 3 * 256 * 256;
    for (int i = 0; i < 6; ++i)
        pack_conv_w<<<(npack + 255) / 256, 256, 0, stream>>>(convw[i], wpk[i]);

    // 2) Duration cumsum + length regulate (h into d_out)
    cumsum_kernel<<<B_, 32, 0, stream>>>(d_target, cum);
    length_regulate_kernel<<<(B_ * T_) / 8, 256, 0, stream>>>(x, cum, h_out);

    auto run_pred = [&](const float* input, int T_len, int p, float* outp) {
        dim3 grid(T_len / TT, B_);
        conv_ln_kernel<float><<<grid, 256, 0, stream>>>(
            input, wpk[p * 2], prm[p][1], prm[p][2], prm[p][3], ln1, T_len);
        conv_ln_kernel<_Float16><<<grid, 256, 0, stream>>>(
            ln1, wpk[p * 2 + 1], prm[p][5], prm[p][6], prm[p][7], ln2, T_len);
        linear_head_kernel<<<(B_ * T_len) / 8, 256, 0, stream>>>(ln2, prm[p][8], prm[p][9], outp);
    };

    // 3) log_dur = predictor(x)
    run_pred(x, L_, 0, log_dur);

    // 4) energy_pred = predictor(h); h += en_emb[bucketize(e_target)]
    run_pred(h_out, T_, 1, energy);
    add_emb_kernel<<<(B_ * T_) / 8, 256, 0, stream>>>(h_out, e_target, en_bins, en_emb);

    // 5) pitch_pred = predictor(h); h += pi_emb[bucketize(p_target)]
    run_pred(h_out, T_, 2, pitch);
    add_emb_kernel<<<(B_ * T_) / 8, 256, 0, stream>>>(h_out, p_target, pi_bins, pi_emb);
}
